// ChatGLMSelfAttention_tp_29334626632457
// MI455X (gfx1250) — compile-verified, matrix operand order checked by reading, not run on hardware
//
#include <hip/hip_runtime.h>
#include <hip/hip_bf16.h>

// Problem constants (from reference)
#define SQ_   2048
#define B_    2
#define H_    4096
#define NH_   32
#define HD_   128
#define NG_   2
#define ROT_  64
#define QKVN_ (NH_*HD_ + 2*NG_*HD_)   // 4608
#define MROWS_ (SQ_*B_)               // 4096

typedef __attribute__((ext_vector_type(16))) __bf16 v16bf;
typedef __attribute__((ext_vector_type(8)))  float  v8f;

union Frag { v16bf v; unsigned u[8]; };

__device__ __forceinline__ v8f wmma_bf16(v16bf a, v16bf b, v8f c) {
  // D(f32 16x16) = A(bf16 16x32) * B(bf16 32x16) + C
  return __builtin_amdgcn_wmma_f32_16x16x32_bf16(
      false, a, false, b, (short)0, c, false, false);
}

// 16-bit A(16x32) / mirrored B(32x16) fragment: pair p (elements 2p,2p+1) sits
// at k = 2p + 8*half + (p>=4 ? 8 : 0)   (k always even -> 32-bit loads; each
// half's 4 pairs are 16B-contiguous -> compiler merges to ds_load_b128)
__device__ __forceinline__ int frag_k(int p, int half) {
  return 2 * p + 8 * half + ((p >= 4) ? 8 : 0);
}

// interleave two rows (k, k+1) of 8 bf16 each into 8 pair-packed uints
__device__ __forceinline__ void interleave_store(unsigned* dst, uint4 x, uint4 y) {
  unsigned xa[4] = {x.x, x.y, x.z, x.w};
  unsigned ya[4] = {y.x, y.y, y.z, y.w};
  unsigned o[8];
#pragma unroll
  for (int j = 0; j < 4; ++j) {
    o[2*j]   = (xa[j] & 0xffffu) | (ya[j] << 16);
    o[2*j+1] = (xa[j] >> 16)     | (ya[j] & 0xffff0000u);
  }
  uint4 lo = {o[0], o[1], o[2], o[3]};
  uint4 hi = {o[4], o[5], o[6], o[7]};
  *(uint4*)(dst)     = lo;
  *(uint4*)(dst + 4) = hi;
}

// ---------------------------------------------------------------------------
// Vectorized f32 -> bf16 conversion (8 elements / thread, b128 in, b128 out)
// ---------------------------------------------------------------------------
__launch_bounds__(256)
__global__ void cvt_f32_bf16(const float* __restrict__ src,
                             __bf16* __restrict__ dst, int n8) {
  int i = blockIdx.x * 256 + threadIdx.x;
  if (i >= n8) return;
  size_t off = (size_t)i * 8;
  float4 a = *(const float4*)&src[off];
  float4 b = *(const float4*)&src[off + 4];
  union { uint4 u; __bf16 h[8]; } o;
  o.h[0] = (__bf16)a.x; o.h[1] = (__bf16)a.y;
  o.h[2] = (__bf16)a.z; o.h[3] = (__bf16)a.w;
  o.h[4] = (__bf16)b.x; o.h[5] = (__bf16)b.y;
  o.h[6] = (__bf16)b.z; o.h[7] = (__bf16)b.w;
  *(uint4*)&dst[off] = o.u;
}

// ---------------------------------------------------------------------------
// bf16 WMMA GEMM:  C[M,N](f32) = A[M,K](bf16) * B[K,N](bf16) (+ bias)
// Block tile 128x256, 256 threads = 8 waves (2x4), wave tile 64x64.
// Double-buffered LDS, software-pipelined global->LDS staging (b128 only).
// ---------------------------------------------------------------------------
__launch_bounds__(256)
__global__ void gemm_bf16_wmma(const __bf16* __restrict__ A,
                               const __bf16* __restrict__ Bm,
                               const float* __restrict__ bias,
                               float* __restrict__ C,
                               int M, int N, int K) {
  __shared__ __bf16  As[2][128][40];     // [m][k], stride 80B (16B aligned)
  __shared__ unsigned Bs2[2][16][256];   // pair-packed: [k/2][n] = {B[k][n],B[k+1][n]}

  const int tid  = threadIdx.x;
  const int lane = tid & 31;
  const int wave = tid >> 5;
  const int half = lane >> 4;
  const int ln   = lane & 15;
  const int wm   = (wave >> 2) * 64;     // 0 / 64
  const int wn   = (wave & 3) * 64;      // 0 / 64 / 128 / 192
  const int m0   = blockIdx.y * 128;
  const int n0   = blockIdx.x * 256;

  // staging geometry
  const int arow = tid >> 1, acol = (tid & 1) * 16;   // A: 1 b128 / thread
  const int bkp  = tid >> 5, bn8 = (tid & 31) * 8;    // B: 2 units / thread

  const v8f vzero = {0.f,0.f,0.f,0.f,0.f,0.f,0.f,0.f};
  v8f acc[4][4];
#pragma unroll
  for (int mt = 0; mt < 4; ++mt)
#pragma unroll
    for (int nt = 0; nt < 4; ++nt) acc[mt][nt] = vzero;

  uint4 ar, bx[2], by[2];
  auto gload = [&](int k0) {
    ar = *(const uint4*)&A[(size_t)(m0 + arow) * K + k0 + acol];
#pragma unroll
    for (int i = 0; i < 2; ++i) {
      int kp = bkp + 8 * i;
      bx[i] = *(const uint4*)&Bm[(size_t)(k0 + 2*kp)     * N + n0 + bn8];
      by[i] = *(const uint4*)&Bm[(size_t)(k0 + 2*kp + 1) * N + n0 + bn8];
    }
  };
  auto sstore = [&](int buf) {
    *(uint4*)&As[buf][arow][acol] = ar;
#pragma unroll
    for (int i = 0; i < 2; ++i)
      interleave_store(&Bs2[buf][bkp + 8*i][bn8], bx[i], by[i]);
  };

  gload(0);
  sstore(0);
  int buf = 0;

  for (int k0 = 0; k0 < K; k0 += 32) {
    __syncthreads();
    const bool more = (k0 + 32) < K;
    if (more) gload(k0 + 32);                       // prefetch next tile
    if (k0 + 64 < K) {                              // L2 prefetch hint
      __builtin_prefetch(&A[(size_t)(m0 + arow) * K + k0 + 64 + acol], 0, 1);
      __builtin_prefetch(&Bm[(size_t)(k0 + 64 + 2*bkp) * N + n0 + bn8], 0, 1);
    }

    // A fragments (4 m-tiles)
    Frag af[4];
#pragma unroll
    for (int mt = 0; mt < 4; ++mt) {
      int r = wm + mt * 16 + ln;
#pragma unroll
      for (int p = 0; p < 8; ++p)
        af[mt].u[p] = *(const unsigned*)&As[buf][r][frag_k(p, half)];
    }
    // B fragments loaded per n-tile (keeps register pressure down)
#pragma unroll
    for (int nt = 0; nt < 4; ++nt) {
      int n = wn + nt * 16 + ln;
      Frag bfr;
#pragma unroll
      for (int p = 0; p < 8; ++p)
        bfr.u[p] = Bs2[buf][frag_k(p, half) >> 1][n];
#pragma unroll
      for (int mt = 0; mt < 4; ++mt)
        acc[mt][nt] = wmma_bf16(af[mt].v, bfr.v, acc[mt][nt]);
    }

    if (more) sstore(buf ^ 1);
    buf ^= 1;
  }

  // epilogue: lane ln = col, VGPR r -> row (r + 8*half)
#pragma unroll
  for (int mt = 0; mt < 4; ++mt)
#pragma unroll
    for (int nt = 0; nt < 4; ++nt) {
      int col = n0 + wn + nt * 16 + ln;
      float bv = bias ? bias[col] : 0.f;
#pragma unroll
      for (int r = 0; r < 8; ++r) {
        int row = m0 + wm + mt * 16 + r + 8 * half;
        C[(size_t)row * N + col] = acc[mt][nt][r] + bv;
      }
    }
}

// ---------------------------------------------------------------------------
// RoPE + bf16 repack. Q is pre-scaled by 1/sqrt(HD) (folds softmax scale).
//   Q -> [b][h][s][d],  K -> [b][g][t][d],  V -> [b][g][t][d]
// ---------------------------------------------------------------------------
__launch_bounds__(256)
__global__ void rotary_pack(const float* __restrict__ mixed,
                            const float* __restrict__ rope,
                            __bf16* __restrict__ Qb,
                            __bf16* __restrict__ Kb,
                            __bf16* __restrict__ Vb) {
  const int s = blockIdx.x;
  const int b = blockIdx.y;
  const float* row = mixed + ((size_t)s * B_ + b) * QKVN_;
  const float* rp  = rope + (size_t)s * ROT_;
  const float qscale = 0.08838834764831845f;   // 1/sqrt(128)

  const int QU = NH_ * (HD_ / 2);
  const int KU = NG_ * (HD_ / 2);
  const int VU = NG_ * (HD_ / 2);

  for (int u = threadIdx.x; u < QU + KU + VU; u += blockDim.x) {
    if (u < QU) {
      int h = u >> 6, j = u & 63, d = 2 * j;
      float x0 = row[h * HD_ + d], x1 = row[h * HD_ + d + 1];
      if (d < ROT_) {
        float c = rp[2*j], sn = rp[2*j + 1];
        float y0 = x0 * c - x1 * sn, y1 = x1 * c + x0 * sn;
        x0 = y0; x1 = y1;
      }
      x0 *= qscale; x1 *= qscale;
      size_t dst = (((size_t)b * NH_ + h) * SQ_ + s) * HD_ + d;
      Qb[dst] = (__bf16)x0; Qb[dst + 1] = (__bf16)x1;
    } else if (u < QU + KU) {
      int uu = u - QU, g = uu >> 6, j = uu & 63, d = 2 * j;
      float x0 = row[NH_*HD_ + g*HD_ + d];
      float x1 = row[NH_*HD_ + g*HD_ + d + 1];
      if (d < ROT_) {
        float c = rp[2*j], sn = rp[2*j + 1];
        float y0 = x0 * c - x1 * sn, y1 = x1 * c + x0 * sn;
        x0 = y0; x1 = y1;
      }
      size_t dst = (((size_t)b * NG_ + g) * SQ_ + s) * HD_ + d;
      Kb[dst] = (__bf16)x0; Kb[dst + 1] = (__bf16)x1;
    } else {
      int uu = u - QU - KU, g = uu >> 6, j = uu & 63, d = 2 * j;
      float x0 = row[NH_*HD_ + NG_*HD_ + g*HD_ + d];
      float x1 = row[NH_*HD_ + NG_*HD_ + g*HD_ + d + 1];
      size_t dst = (((size_t)b * NG_ + g) * SQ_ + s) * HD_ + d;
      Vb[dst] = (__bf16)x0; Vb[dst + 1] = (__bf16)x1;
    }
  }
}

// ---------------------------------------------------------------------------
// Causal flash attention (GQA). grid = (SQ/128, B*NH), block = 256 (8 waves).
// 128 q-rows per block: K/V staged once per 128 rows (2x less staging traffic
// than 64-row blocks). Wave w owns q-rows [qb*128+16w, +16). Mask code runs
// only for waves whose rows intersect the diagonal. ctx written as bf16.
// ---------------------------------------------------------------------------
__launch_bounds__(256)
__global__ void flash_attn(const __bf16* __restrict__ Qb,
                           const __bf16* __restrict__ Kb,
                           const __bf16* __restrict__ Vb,
                           __bf16* __restrict__ ctxb) {
  __shared__ __bf16  Ks[64][HD_ + 8];    // [t][d], stride 272B (16B aligned)
  __shared__ unsigned Vp[32][HD_];       // pair-packed over t: [t/2][d]
  __shared__ __bf16  Ps[128][64 + 2];    // [s_local][t]

  const int tid  = threadIdx.x;
  const int lane = tid & 31;
  const int wave = tid >> 5;
  const int half = lane >> 4;
  const int ln   = lane & 15;
  const int qb   = blockIdx.x;           // 128-row query block
  const int bh   = blockIdx.y;
  const int b    = bh / NH_;
  const int h    = bh % NH_;
  const int g    = h / (NH_ / NG_);

  // Q fragments (pre-scaled at RoPE time), 4 chunks over d
  const __bf16* qr =
      Qb + (((size_t)b * NH_ + h) * SQ_ + (qb * 128 + wave * 16 + ln)) * HD_;
  Frag qf[4];
#pragma unroll
  for (int kc = 0; kc < 4; ++kc)
#pragma unroll
    for (int p = 0; p < 8; ++p)
      qf[kc].u[p] = *(const unsigned*)&qr[kc * 32 + frag_k(p, half)];

  const __bf16* kbase = Kb + ((size_t)b * NG_ + g) * SQ_ * HD_;
  const __bf16* vbase = Vb + ((size_t)b * NG_ + g) * SQ_ * HD_;

  const v8f vzero = {0.f,0.f,0.f,0.f,0.f,0.f,0.f,0.f};
  v8f o[8];
#pragma unroll
  for (int dt = 0; dt < 8; ++dt) o[dt] = vzero;
  float mrow[8], lrow[8];
#pragma unroll
  for (int r = 0; r < 8; ++r) { mrow[r] = -1e30f; lrow[r] = 0.f; }

  const int kt_max = 2 * qb + 1;         // causal: key blocks of 64
  for (int kt = 0; kt <= kt_max; ++kt) {
    // ---- stage K (row-major b128 copies): 1024 chunks of 8 bf16 ----
#pragma unroll
    for (int i = 0; i < 4; ++i) {
      int idx = tid + i * 256;
      int t = idx >> 4, d8 = (idx & 15) * 8;
      *(uint4*)&Ks[t][d8] =
          *(const uint4*)&kbase[(size_t)(kt * 64 + t) * HD_ + d8];
    }
    // ---- stage V pair-packed over t: 512 units ----
#pragma unroll
    for (int i = 0; i < 2; ++i) {
      int u = tid + i * 256;
      int tp = u >> 4, d8 = (u & 15) * 8;
      uint4 x = *(const uint4*)&vbase[(size_t)(kt*64 + 2*tp)     * HD_ + d8];
      uint4 y = *(const uint4*)&vbase[(size_t)(kt*64 + 2*tp + 1) * HD_ + d8];
      interleave_store(&Vp[tp][d8], x, y);
    }
    if (kt < kt_max)                                  // L2 prefetch hint
      __builtin_prefetch(&kbase[(size_t)((kt + 1) * 64) * HD_ + tid * 8], 0, 1);
    __syncthreads();

    // ---- S = Q * K^T : 4 n-tiles (t) x 4 k-chunks (d) ----
    v8f sacc[4];
#pragma unroll
    for (int nt = 0; nt < 4; ++nt) {
      sacc[nt] = vzero;
      int t = nt * 16 + ln;
#pragma unroll
      for (int kc = 0; kc < 4; ++kc) {
        Frag kf;
#pragma unroll
        for (int p = 0; p < 8; ++p)
          kf.u[p] = *(const unsigned*)&Ks[t][kc * 32 + frag_k(p, half)];
        sacc[nt] = wmma_bf16(qf[kc].v, kf.v, sacc[nt]);
      }
    }

    // ---- causal mask only where this wave's rows meet the diagonal ----
    const int s_min = qb * 128 + wave * 16;
    if (kt * 64 + 63 > s_min) {
      const int s_base = s_min + 8 * half;
#pragma unroll
      for (int nt = 0; nt < 4; ++nt) {
        int t = kt * 64 + nt * 16 + ln;
#pragma unroll
        for (int r = 0; r < 8; ++r)
          if (t > s_base + r) sacc[nt][r] = -1e30f;
      }
    }

    // ---- online softmax (row r+8*half lives in one 16-lane half) ----
#pragma unroll
    for (int r = 0; r < 8; ++r) {
      float mx = -1e30f;
#pragma unroll
      for (int nt = 0; nt < 4; ++nt) mx = fmaxf(mx, sacc[nt][r]);
#pragma unroll
      for (int off = 8; off >= 1; off >>= 1)
        mx = fmaxf(mx, __shfl_xor(mx, off, 32));
      float mnew  = fmaxf(mrow[r], mx);
      float alpha = __expf(mrow[r] - mnew);
      float psum = 0.f;
#pragma unroll
      for (int nt = 0; nt < 4; ++nt) {
        float p = __expf(sacc[nt][r] - mnew);
        sacc[nt][r] = p;
        psum += p;
      }
#pragma unroll
      for (int off = 8; off >= 1; off >>= 1)
        psum += __shfl_xor(psum, off, 32);
      lrow[r] = lrow[r] * alpha + psum;
      mrow[r] = mnew;
#pragma unroll
      for (int dt = 0; dt < 8; ++dt) o[dt][r] *= alpha;
    }

    // ---- P -> LDS (bf16) ----
#pragma unroll
    for (int nt = 0; nt < 4; ++nt)
#pragma unroll
      for (int r = 0; r < 8; ++r)
        Ps[wave * 16 + r + 8 * half][nt * 16 + ln] = (__bf16)sacc[nt][r];
    __syncthreads();

    // ---- O += P * V ----
    Frag pf[2];
#pragma unroll
    for (int kc = 0; kc < 2; ++kc)
#pragma unroll
      for (int p = 0; p < 8; ++p)
        pf[kc].u[p] =
            *(const unsigned*)&Ps[wave * 16 + ln][kc * 32 + frag_k(p, half)];
#pragma unroll
    for (int dt = 0; dt < 8; ++dt) {
      int d = dt * 16 + ln;
#pragma unroll
      for (int kc = 0; kc < 2; ++kc) {
        Frag vf;
#pragma unroll
        for (int p = 0; p < 8; ++p)
          vf.u[p] = Vp[kc * 16 + (frag_k(p, half) >> 1)][d];
        o[dt] = wmma_bf16(pf[kc].v, vf.v, o[dt]);
      }
    }
    __syncthreads();
  }

  // ---- epilogue: ctx_bf16[(s*B + b)][h*HD + d] = O / l ----
#pragma unroll
  for (int dt = 0; dt < 8; ++dt) {
    int d = dt * 16 + ln;
#pragma unroll
    for (int r = 0; r < 8; ++r) {
      int s = qb * 128 + wave * 16 + r + 8 * half;
      ctxb[((size_t)s * B_ + b) * (NH_ * HD_) + h * HD_ + d] =
          (__bf16)(o[dt][r] / lrow[r]);
    }
  }
}

// ---------------------------------------------------------------------------
extern "C" void kernel_launch(void* const* d_in, const int* in_sizes, int n_in,
                              void* d_out, int out_size, void* d_ws, size_t ws_size,
                              hipStream_t stream) {
  const float* hidden = (const float*)d_in[0];
  // d_in[1] = attention_mask (causal triu, k=1) -> implied, unused
  const float* rope   = (const float*)d_in[2];
  const float* Wqkv   = (const float*)d_in[3];
  const float* bqkv   = (const float*)d_in[4];
  const float* Wd     = (const float*)d_in[5];
  float* out = (float*)d_out;

  // Workspace layout (regions reused across stream-ordered kernels):
  //  R0: mixed f32 (75.5MB) -- ctx_bf16 (33.5MB) overlays it after rotary
  //  R1: hidden_bf16 (33.5MB) -- Wd_bf16 overlays it after gemm1
  //  R2: Wqkv_bf16 (37.7MB);  then Qb, Kb, Vb
  char* ws = (char*)d_ws;
  float*  mixed = (float*)ws;
  __bf16* ctxb  = (__bf16*)ws;
  ws += (size_t)MROWS_ * QKVN_ * sizeof(float);
  __bf16* hidb  = (__bf16*)ws;
  __bf16* Wdb   = (__bf16*)ws;
  ws += (size_t)MROWS_ * H_ * sizeof(__bf16);
  __bf16* Wqkvb = (__bf16*)ws;  ws += (size_t)H_ * QKVN_ * sizeof(__bf16);
  __bf16* Qb = (__bf16*)ws;  ws += (size_t)B_ * NH_ * SQ_ * HD_ * sizeof(__bf16);
  __bf16* Kb = (__bf16*)ws;  ws += (size_t)B_ * NG_ * SQ_ * HD_ * sizeof(__bf16);
  __bf16* Vb = (__bf16*)ws;  ws += (size_t)B_ * NG_ * SQ_ * HD_ * sizeof(__bf16);

  // 0) f32 -> bf16 conversions
  {
    int n8 = (MROWS_ * H_) / 8;
    cvt_f32_bf16<<<n8 / 256, 256, 0, stream>>>(hidden, hidb, n8);
  }
  {
    int n8 = (H_ * QKVN_) / 8;
    cvt_f32_bf16<<<n8 / 256, 256, 0, stream>>>(Wqkv, Wqkvb, n8);
  }

  // 1) QKV projection (+bias): [4096x4096]x[4096x4608] -> mixed (f32)
  gemm_bf16_wmma<<<dim3(QKVN_ / 256, MROWS_ / 128), 256, 0, stream>>>(
      hidb, Wqkvb, bqkv, mixed, MROWS_, QKVN_, H_);

  // 2) RoPE + bf16 repack (Q pre-scaled)
  rotary_pack<<<dim3(SQ_, B_), 256, 0, stream>>>(mixed, rope, Qb, Kb, Vb);

  // 3) Causal flash attention -> ctx (bf16; overlays mixed)
  flash_attn<<<dim3(SQ_ / 128, B_ * NH_), 256, 0, stream>>>(Qb, Kb, Vb, ctxb);

  // 4) Convert W_dense (overlays hidden_bf16), then dense projection -> out
  {
    int n8 = (H_ * H_) / 8;
    cvt_f32_bf16<<<n8 / 256, 256, 0, stream>>>(Wd, Wdb, n8);
  }
  gemm_bf16_wmma<<<dim3(H_ / 256, MROWS_ / 128), 256, 0, stream>>>(
      ctxb, Wdb, nullptr, out, MROWS_, H_, H_);
}
